// DeformableSliceGrouped_20950850470413
// MI455X (gfx1250) — matrix operand order repaired and checked by reading, not compiled, not run on hardware
//
#include <hip/hip_runtime.h>
#include <hip/hip_bf16.h>
#include <math.h>

// ---------- types ----------
typedef __attribute__((ext_vector_type(8)))  float  v8f;
typedef __attribute__((ext_vector_type(8)))  __bf16 v8bf;
typedef __attribute__((ext_vector_type(16))) __bf16 v16bf;

// Problem constants (from reference setup_inputs)
#define B_   2
#define C_   256
#define ZS_  32
#define HW_  2304          // 48*48
#define NSLICE (B_*ZS_)    // 64
#define HP_  6             // N_HEADS*N_POINTS
#define NRED (ZS_*HW_)     // per-channel per-batch elems = 73728
#define NPART 18           // 2 b * 9 hw-chunks partials per channel

static __device__ __forceinline__ __bf16 f2bf(float f) {
    union { float f; unsigned u; } v{f};
    // round-to-nearest-even bf16
    unsigned r = (v.u + 0x7FFFu + ((v.u >> 16) & 1u)) >> 16;
    unsigned short h = (unsigned short)r;
    __bf16 b;
    __builtin_memcpy(&b, &h, 2);
    return b;
}

static __device__ __forceinline__ v16bf cat16(v8bf lo, v8bf hi) {
    v16bf r;
#pragma unroll
    for (int i = 0; i < 8; i++) { r[i] = lo[i]; r[i + 8] = hi[i]; }
    return r;
}

// orderable encoding of float for atomicMax on unsigned (0 == -inf)
static __device__ __forceinline__ unsigned encf(float f) {
    unsigned b = __float_as_uint(f);
    return (b & 0x80000000u) ? ~b : (b | 0x80000000u);
}
static __device__ __forceinline__ float decf(unsigned e) {
    unsigned b = (e & 0x80000000u) ? (e ^ 0x80000000u) : ~e;
    return __uint_as_float(b);
}

// ---------------------------------------------------------------------------
// kPrep: W = o_w @ v_w  -> bf16 ;  q_w -> bf16 ; pe table ; zero qp encodings
// grid(256) x block(256): block m computes row m of W (k contiguous ✓ A-frag)
// ---------------------------------------------------------------------------
__global__ __launch_bounds__(256) void kPrep(const float* __restrict__ qw,
                                             const float* __restrict__ vw,
                                             const float* __restrict__ ow,
                                             __bf16* __restrict__ wqb,
                                             __bf16* __restrict__ wob,
                                             float* __restrict__ peB,
                                             unsigned* __restrict__ qpE) {
    const int m = blockIdx.x, n = threadIdx.x;
    float s = 0.f;
    for (int c = 0; c < C_; c++)
        s = fmaf(ow[m * C_ + c], vw[c * C_ + n], s);   // W[o,k]=sum_c o_w[o,c]*v_w[c,k]
    wob[m * C_ + n] = f2bf(s);
    wqb[m * C_ + n] = f2bf(qw[m * C_ + n]);
    if (m < NSLICE) qpE[m * C_ + n] = 0u;              // -inf encoded
    if (m < ZS_) {
        const int k = n;
        const int i2 = k & ~1;                          // even channel pair base (2i)
        const float div = __expf(-(float)i2 * (logf(10000.f) / (float)C_));
        const float ang = (float)m * div;
        peB[m * C_ + k] = (k & 1) ? __cosf(ang) : __sinf(ang);
    }
}

// ---------------------------------------------------------------------------
// kGemm: per (b,z) slice, q = q_w@(x+pe) (spatial-max only) and u = W@(x+pe)
// bf16 WMMA 16x16x32, f32 accumulate. grid(36 Nblk, 2 Mblk, 64 slices), 256 thr.
// Each wave owns a 16(M)x64(N) strip: 4 N-tiles x 2 matrices, shared B frag.
// ---------------------------------------------------------------------------
__global__ __launch_bounds__(256) void kGemm(const float* __restrict__ feat,
                                             const __bf16* __restrict__ wqb,
                                             const __bf16* __restrict__ wob,
                                             const float* __restrict__ peB,
                                             float* __restrict__ u,
                                             unsigned* __restrict__ qpE) {
    __shared__ __bf16 xs[64 * 40];      // [n][k] bf16, stride 40 halves (pad: no bank conflicts)

    const int tid  = threadIdx.x;
    const int wave = tid >> 5, lane = tid & 31;
    const int g    = lane >> 4, ml = lane & 15;
    const int n0    = blockIdx.x * 64;
    const int mblk  = blockIdx.y;
    const int slice = blockIdx.z;
    const int b = slice >> 5, z = slice & 31;
    const int m = mblk * 128 + wave * 16 + ml;          // A-matrix row for this lane

    v8f accq[4]; v8f accu[4];
#pragma unroll
    for (int t = 0; t < 4; t++) { accq[t] = (v8f)0.f; accu[t] = (v8f)0.f; }

    const int ln  = tid & 63;          // hw column inside tile
    const int lk0 = (tid >> 6) * 8;    // this thread's K sub-range for staging
    const size_t fbase = ((size_t)b * C_) * (size_t)(ZS_ * HW_)
                       + (size_t)z * HW_ + (size_t)(n0 + ln);

    for (int k0 = 0; k0 < C_; k0 += 32) {
        __syncthreads();
        // stage x tile (32K x 64N), add PE, convert to bf16
#pragma unroll
        for (int kk = 0; kk < 8; kk++) {
            const int k = k0 + lk0 + kk;
            float f = feat[fbase + (size_t)k * (ZS_ * HW_)] + peB[z * C_ + k];
            xs[ln * 40 + lk0 + kk] = f2bf(f);
        }
        if (k0 < C_ - 32)   // gfx1250 global_prefetch_b8 for the next K panel
            __builtin_prefetch(&feat[fbase + (size_t)(k0 + 32 + lk0) * (ZS_ * HW_)], 0, 1);
        __syncthreads();

        // A fragments: 16-bit A 16x32 layout -> halves = K in [8g,8g+8) U [16+8g,24+8g)
        const v16bf aq = cat16(*(const v8bf*)&wqb[(size_t)m * C_ + k0 + 8 * g],
                               *(const v8bf*)&wqb[(size_t)m * C_ + k0 + 16 + 8 * g]);
        const v16bf au = cat16(*(const v8bf*)&wob[(size_t)m * C_ + k0 + 8 * g],
                               *(const v8bf*)&wob[(size_t)m * C_ + k0 + 16 + 8 * g]);
#pragma unroll
        for (int nt = 0; nt < 4; nt++) {
            // B fragment: N = lane&15, halves = K [8g,8g+8) U [16+8g,24+8g)
            const int nb = (nt * 16 + ml) * 40 + 8 * g;
            const v16bf bf = cat16(*(const v8bf*)&xs[nb],
                                   *(const v8bf*)&xs[nb + 16]);
            accq[nt] = __builtin_amdgcn_wmma_f32_16x16x32_bf16(
                false, aq, false, bf, (short)0, accq[nt], false, false);
            accu[nt] = __builtin_amdgcn_wmma_f32_16x16x32_bf16(
                false, au, false, bf, (short)0, accu[nt], false, false);
        }
    }

    // ---- write u (f32), C/D layout: M = r + 8g, N = ml ----
    const int mrow = mblk * 128 + wave * 16 + 8 * g;
#pragma unroll
    for (int nt = 0; nt < 4; nt++) {
        const int col = n0 + nt * 16 + ml;
#pragma unroll
        for (int r = 0; r < 8; r++) {
            u[(((size_t)b * C_ + (mrow + r)) * ZS_ + z) * HW_ + col] = accu[nt][r];
        }
    }

    // ---- spatial max of q: reduce over this wave's 64 columns, then atomic ----
    v8f mq = accq[0];
#pragma unroll
    for (int nt = 1; nt < 4; nt++)
#pragma unroll
        for (int r = 0; r < 8; r++) mq[r] = fmaxf(mq[r], accq[nt][r]);
#pragma unroll
    for (int off = 1; off < 16; off <<= 1)
#pragma unroll
        for (int r = 0; r < 8; r++)
            mq[r] = fmaxf(mq[r], __shfl_xor(mq[r], off, 16));  // within 16-lane group
    if (ml == 0) {
#pragma unroll
        for (int r = 0; r < 8; r++)
            atomicMax(&qpE[slice * C_ + mrow + r], encf(mq[r]));
    }
}

// ---------------------------------------------------------------------------
// kBuildM: per (b,z) row -> offsets, softmax attn, 32x32 mixing matrix M[b]
// grid(64) x block(64); tiny.
// ---------------------------------------------------------------------------
__global__ void kBuildM(const unsigned* __restrict__ qpE,
                        const float* __restrict__ offsW, const float* __restrict__ offsB,
                        const float* __restrict__ attnW, const float* __restrict__ attnB,
                        float* __restrict__ Mmat) {
    __shared__ float off_s[HP_], t_s[HP_];
    const int row = blockIdx.x, tid = threadIdx.x;
    if (tid < HP_) {
        float s = offsB[tid], t = attnB[tid];
        for (int c = 0; c < C_; c++) {
            const float q = decf(qpE[row * C_ + c]);
            s = fmaf(offsW[tid * C_ + c], q, s);
            t = fmaf(attnW[tid * C_ + c], q, t);
        }
        off_s[tid] = fminf(fmaxf(s, 0.f), (float)(ZS_ - 1));
        t_s[tid] = t;
    }
    __syncthreads();
    if (tid == 0) {
        float mt = -1e30f;
        for (int j = 0; j < HP_; j++) mt = fmaxf(mt, t_s[j]);
        float e[HP_], se = 0.f;
        for (int j = 0; j < HP_; j++) { e[j] = __expf(t_s[j] - mt); se += e[j]; }
        float Mrow[ZS_];
        for (int y = 0; y < ZS_; y++) Mrow[y] = 0.f;
        for (int j = 0; j < HP_; j++) {
            const float a = e[j] / se;
            const float lo = floorf(off_s[j]);
            const float hi = ceilf(off_s[j]);
            const float fr = off_s[j] - lo;
            Mrow[(int)lo] += a * (1.f - fr);
            Mrow[(int)hi] += a * fr;
        }
        for (int y = 0; y < ZS_; y++) Mmat[row * ZS_ + y] = Mrow[y];
    }
}

// ---------------------------------------------------------------------------
// kMix: out_pre[b,c,z,hw] = sum_y M[b,z,y] * u[b,c,y,hw]; partial BN sums.
// grid(9 hw-chunks, 256 c, 2 b) x block(256). Deterministic (no FP atomics).
// ---------------------------------------------------------------------------
__global__ __launch_bounds__(256) void kMix(const float* __restrict__ u,
                                            const float* __restrict__ Mmat,
                                            float* __restrict__ out,
                                            float* __restrict__ psum,
                                            float* __restrict__ psum2) {
    __shared__ float Ms[ZS_ * ZS_];
    __shared__ float r1[256], r2[256];
    const int chunk = blockIdx.x, c = blockIdx.y, b = blockIdx.z, tid = threadIdx.x;
    for (int j = tid; j < ZS_ * ZS_; j += 256) Ms[j] = Mmat[b * ZS_ * ZS_ + j];
    __syncthreads();

    const int hw = chunk * 256 + tid;
    const size_t base = (((size_t)b * C_ + c) * ZS_) * HW_ + hw;
    float uy[ZS_];
#pragma unroll
    for (int y = 0; y < ZS_; y++) uy[y] = u[base + (size_t)y * HW_];

    float s = 0.f, s2 = 0.f;
    for (int z = 0; z < ZS_; z++) {
        float o = 0.f;
#pragma unroll
        for (int y = 0; y < ZS_; y++) o = fmaf(Ms[z * ZS_ + y], uy[y], o);
        out[base + (size_t)z * HW_] = o;
        s += o; s2 += o * o;
    }
    r1[tid] = s; r2[tid] = s2;
    __syncthreads();
    for (int st = 128; st > 0; st >>= 1) {
        if (tid < st) { r1[tid] += r1[tid + st]; r2[tid] += r2[tid + st]; }
        __syncthreads();
    }
    if (tid == 0) {
        psum [c * NPART + b * 9 + chunk] = r1[0];
        psum2[c * NPART + b * 9 + chunk] = r2[0];
    }
}

// ---------------------------------------------------------------------------
// kStats: fold partials -> per-channel scale/shift for BN (+gamma/beta fold)
// ---------------------------------------------------------------------------
__global__ void kStats(const float* __restrict__ psum, const float* __restrict__ psum2,
                       const float* __restrict__ gamma, const float* __restrict__ beta,
                       float* __restrict__ scale, float* __restrict__ shift) {
    const int c = threadIdx.x;
    float s = 0.f, s2 = 0.f;
    for (int j = 0; j < NPART; j++) { s += psum[c * NPART + j]; s2 += psum2[c * NPART + j]; }
    const float inv_n = 1.f / (float)(B_ * NRED);
    const float mean = s * inv_n;
    const float var  = s2 * inv_n - mean * mean;        // biased var (matches jnp.var)
    const float sc   = gamma[c] * rsqrtf(var + 1e-5f);
    scale[c] = sc;
    shift[c] = beta[c] - mean * sc;
}

// ---------------------------------------------------------------------------
// kFinal: out = out*scale[c] + shift[c] + features   (float4 vectorized)
// ---------------------------------------------------------------------------
__global__ __launch_bounds__(256) void kFinal(float* __restrict__ out,
                                              const float* __restrict__ feat,
                                              const float* __restrict__ scale,
                                              const float* __restrict__ shift) {
    const size_t i4 = (size_t)blockIdx.x * blockDim.x + threadIdx.x;  // float4 index
    const int c = (int)((i4 * 4 / (size_t)(ZS_ * HW_)) % C_);
    const float sc = scale[c], sh = shift[c];
    float4 p = ((const float4*)out)[i4];
    const float4 f = ((const float4*)feat)[i4];
    p.x = fmaf(p.x, sc, sh) + f.x;
    p.y = fmaf(p.y, sc, sh) + f.y;
    p.z = fmaf(p.z, sc, sh) + f.z;
    p.w = fmaf(p.w, sc, sh) + f.w;
    ((float4*)out)[i4] = p;
}

// ---------------------------------------------------------------------------
extern "C" void kernel_launch(void* const* d_in, const int* in_sizes, int n_in,
                              void* d_out, int out_size, void* d_ws, size_t ws_size,
                              hipStream_t stream) {
    (void)in_sizes; (void)n_in; (void)out_size; (void)ws_size;
    const float* feat  = (const float*)d_in[0];
    const float* qw    = (const float*)d_in[1];
    const float* vw    = (const float*)d_in[2];
    const float* ow    = (const float*)d_in[3];
    const float* offsW = (const float*)d_in[4];
    const float* offsB = (const float*)d_in[5];
    const float* attnW = (const float*)d_in[6];
    const float* attnB = (const float*)d_in[7];
    const float* gamma = (const float*)d_in[8];
    const float* beta  = (const float*)d_in[9];
    float* out = (float*)d_out;

    char* ws = (char*)d_ws;
    size_t off = 0;
    float*    u     = (float*)(ws + off);    off += (size_t)B_ * C_ * ZS_ * HW_ * 4; // 144 MiB
    __bf16*   wqb   = (__bf16*)(ws + off);   off += (size_t)C_ * C_ * 2;
    __bf16*   wob   = (__bf16*)(ws + off);   off += (size_t)C_ * C_ * 2;
    unsigned* qpE   = (unsigned*)(ws + off); off += (size_t)NSLICE * C_ * 4;
    float*    peB   = (float*)(ws + off);    off += (size_t)ZS_ * C_ * 4;
    float*    Mmat  = (float*)(ws + off);    off += (size_t)B_ * ZS_ * ZS_ * 4;
    float*    psum  = (float*)(ws + off);    off += (size_t)C_ * NPART * 4;
    float*    psum2 = (float*)(ws + off);    off += (size_t)C_ * NPART * 4;
    float*    scale = (float*)(ws + off);    off += (size_t)C_ * 4;
    float*    shift = (float*)(ws + off);    off += (size_t)C_ * 4;

    kPrep  <<<256, 256, 0, stream>>>(qw, vw, ow, wqb, wob, peB, qpE);
    kGemm  <<<dim3(HW_ / 64, 2, NSLICE), 256, 0, stream>>>(feat, wqb, wob, peB, u, qpE);
    kBuildM<<<NSLICE, 64, 0, stream>>>(qpE, offsW, offsB, attnW, attnB, Mmat);
    kMix   <<<dim3(9, C_, B_), 256, 0, stream>>>(u, Mmat, out, psum, psum2);
    kStats <<<1, 256, 0, stream>>>(psum, psum2, gamma, beta, scale, shift);
    kFinal <<<(B_ * C_ * ZS_ * HW_ / 4 + 255) / 256, 256, 0, stream>>>(out, feat, scale, shift);
}
